// ComENet_15942918603431
// MI455X (gfx1250) — matrix operand order repaired and checked by reading, not compiled, not run on hardware
//
#include <hip/hip_runtime.h>
#include <math.h>

typedef __attribute__((ext_vector_type(16))) _Float16 v16h;
typedef __attribute__((ext_vector_type(8)))  float    v8f;
typedef _Float16 half_t;

#define NNODE   20000
#define NEDGE   320000
#define PI_F    3.14159265358979323846f
#define CUTOFF_F 8.0f
#define C0_F 0.28209479177387814f
#define C1_F 0.4886025119029199f

// pack buffer tile map (each tile = 512 halves = 32 lanes x v16h)
//  tiles 0..3   : w_l1 (12x64 padded to 32x64)
//  tiles 4..7   : w_g1 (6x64 padded to 32x64)
//  tiles 8..23  : fused Wl (64x128): kslice s in {0,1}, ntile t in 0..7
//  tiles 24..39 : fused Wg (64x128)
//  tiles 40..71 : w_in  (128x128): kslice s in 0..3, ntile t in 0..7
//  tiles 72..103: w_out (128x128)
#define NTILES 104

// ---------------- helpers ----------------
struct F3 { float x, y, z; };
__device__ __forceinline__ F3 mkf3(float a, float b, float c){ F3 r; r.x=a; r.y=b; r.z=c; return r; }
__device__ __forceinline__ F3 sub3(F3 a, F3 b){ return mkf3(a.x-b.x, a.y-b.y, a.z-b.z); }
__device__ __forceinline__ float dot3(F3 a, F3 b){ return a.x*b.x + a.y*b.y + a.z*b.z; }
__device__ __forceinline__ F3 cross3(F3 a, F3 b){
  return mkf3(a.y*b.z - a.z*b.y, a.z*b.x - a.x*b.z, a.x*b.y - a.y*b.x);
}
__device__ __forceinline__ F3 posv(const float* p, int n){ return mkf3(p[3*n], p[3*n+1], p[3*n+2]); }
__device__ __forceinline__ F3 edgevec(const float* p, const int* el, int e){
  int ni = el[2*e], no = el[2*e+1];
  return sub3(posv(p, no), posv(p, ni));
}
__device__ __forceinline__ float siluf(float v){ return v / (1.0f + expf(-v)); }
__device__ __forceinline__ int clampidx(unsigned long long packed){
  unsigned v = (unsigned)packed;
  return (v < NEDGE) ? (int)v : 0;
}
__device__ __forceinline__ v8f zero8(){
  v8f z = {0.f,0.f,0.f,0.f,0.f,0.f,0.f,0.f};
  return z;
}
// A-operand K index for 16-bit 16x32 tile: lane group g = lane>>4, element e in 0..15
__device__ __forceinline__ int amapK(int g, int e){
  return (e < 8) ? (8*g + e) : (8 + 8*g + e);  // e>=8 -> 16 + 8g + (e-8)
}

// async copy of one 16B chunk per lane: global -> LDS (ASYNCcnt-tracked)
__device__ __forceinline__ void async_g2l_b128(unsigned lds_off, unsigned long long gaddr){
  asm volatile("global_load_async_to_lds_b128 %0, %1, off"
               :: "v"(lds_off), "v"(gaddr) : "memory");
}
__device__ __forceinline__ void wait_asynccnt0(){
  asm volatile("s_wait_asynccnt 0" ::: "memory");
}

// ---------------- init ----------------
__global__ void k_init(float* deg_in, float* deg_out,
                       unsigned long long* mo1, unsigned long long* mi1,
                       unsigned long long* mo2, unsigned long long* mi2,
                       float* upd, float* colsum, float* colsumsq)
{
  int i = blockIdx.x * blockDim.x + threadIdx.x;
  if (i < NNODE * 128) upd[i] = 0.0f;
  if (i < NNODE) {
    deg_in[i] = 1.0f; deg_out[i] = 1.0f;
    mo1[i] = ~0ull; mi1[i] = ~0ull; mo2[i] = ~0ull; mi2[i] = ~0ull;
  }
  if (i < 128) { colsum[i] = 0.0f; colsumsq[i] = 0.0f; }
}

// ---------------- degrees ----------------
__global__ void k_deg(const int* el, float* deg_in, float* deg_out)
{
  int e = blockIdx.x * blockDim.x + threadIdx.x;
  if (e >= NEDGE) return;
  atomicAdd(&deg_in[el[2*e]], 1.0f);
  atomicAdd(&deg_out[el[2*e+1]], 1.0f);
}

// ---------------- segmented argmin (packed dist|idx) ----------------
__global__ void k_min1(const int* el, const float* ef,
                       unsigned long long* mo1, unsigned long long* mi1)
{
  int e = blockIdx.x * blockDim.x + threadIdx.x;
  if (e >= NEDGE) return;
  unsigned long long p = (((unsigned long long)__float_as_uint(ef[e])) << 32) | (unsigned)e;
  atomicMin(&mo1[el[2*e+1]], p);
  atomicMin(&mi1[el[2*e]],   p);
}

__global__ void k_min2(const int* el, const float* ef,
                       const unsigned long long* mo1, const unsigned long long* mi1,
                       unsigned long long* mo2, unsigned long long* mi2)
{
  int e = blockIdx.x * blockDim.x + threadIdx.x;
  if (e >= NEDGE) return;
  int ni = el[2*e], no = el[2*e+1];
  float d = ef[e];
  float vo = d + ((unsigned)mo1[no] == (unsigned)e ? CUTOFF_F : 0.0f);
  float vi = d + ((unsigned)mi1[ni] == (unsigned)e ? CUTOFF_F : 0.0f);
  atomicMin(&mo2[no], (((unsigned long long)__float_as_uint(vo)) << 32) | (unsigned)e);
  atomicMin(&mi2[ni], (((unsigned long long)__float_as_uint(vi)) << 32) | (unsigned)e);
}

// ---------------- geometry + basis features ----------------
__global__ void k_geom(const float* pos, const int* el, const float* ef,
                       const unsigned long long* mo1, const unsigned long long* mi1,
                       const unsigned long long* mo2, const unsigned long long* mi2,
                       half_t* featL, half_t* featG)
{
  int e = blockIdx.x * blockDim.x + threadIdx.x;
  if (e >= NEDGE) return;
  int i = el[2*e], j = el[2*e+1];
  float dist = ef[e];
  F3 pji = sub3(posv(pos, j), posv(pos, i));

  int sf  = clampidx(mo1[j]);
  int ss  = clampidx(mo2[j]);
  int tfj = clampidx(mi1[j]);
  int tfi = clampidx(mi1[i]);
  int tsi = clampidx(mi2[i]);

  int snf = el[2*sf];          // node_in[src_f[j]]
  int idx_iref = (snf == i) ? ss : sf;
  int tnf = el[2*tfi + 1];     // node_out[tgt_f[i]]
  int idx_jref = (tnf == j) ? tsi : tfi;

  F3 pif   = edgevec(pos, el, sf);
  F3 pis   = edgevec(pos, el, tfj);
  F3 piref = edgevec(pos, el, idx_iref);
  F3 pjref = edgevec(pos, el, idx_jref);
  F3 nji = mkf3(-pji.x, -pji.y, -pji.z);

  // theta
  float a = dot3(nji, pif);
  F3 cr = cross3(nji, pif);
  float b = sqrtf(dot3(cr, cr));
  float theta = atan2f(b, a);
  if (theta < 0.0f) theta += PI_F;

  float dji = sqrtf(dot3(pji, pji));

  // phi
  F3 p1 = cross3(nji, pif);
  F3 p2 = cross3(nji, pis);
  a = dot3(p1, p2);
  b = dot3(cross3(p1, p2), pji) / dji;
  float phi = atan2f(b, a);
  if (phi < 0.0f) phi += PI_F;

  // tau
  F3 q1 = cross3(pji, pjref);
  F3 q2 = cross3(pji, piref);
  a = dot3(q1, q2);
  b = dot3(cross3(q1, q2), pji) / dji;
  float tau = atan2f(b, a);
  if (tau < 0.0f) tau += PI_F;

  // spherical Bessel radial basis
  const float Z0c[3] = {PI_F, 2.0f*PI_F, 3.0f*PI_F};
  const float Z1c[3] = {4.493409457909064f, 7.725251836937707f, 10.904121659428899f};
  const float N0c[3] = {4.442882938158366f, 8.885765876316732f, 13.328648814475098f};
  const float N1c[3] = {6.510104f, 11.016313f, 15.485470f};
  float x = dist * (1.0f / CUTOFF_F);
  float b0[3], b1[3];
  #pragma unroll
  for (int k = 0; k < 3; ++k) {
    float t0 = Z0c[k] * x;
    b0[k] = N0c[k] * sinf(t0) / t0;
    float t1 = Z1c[k] * x;
    b1[k] = N1c[k] * (sinf(t1) / (t1*t1) - cosf(t1) / t1);
  }
  float st = sinf(theta), ct = cosf(theta);
  float sp = sinf(phi),  cp = cosf(phi);
  float ctau = cosf(tau);
  float s0 = C0_F, s1 = C1_F*st*sp, s2 = C1_F*ct, s3 = C1_F*st*cp;

  half_t* fl = featL + (size_t)e * 32;
  half_t* fg = featG + (size_t)e * 32;
  #pragma unroll
  for (int k = 0; k < 3; ++k) {
    fl[0 + k] = (half_t)(b0[k] * s0);
    fl[3 + k] = (half_t)(b0[k] * s1);
    fl[6 + k] = (half_t)(b1[k] * s2);
    fl[9 + k] = (half_t)(b1[k] * s3);
    fg[0 + k] = (half_t)(b0[k] * C0_F);
    fg[3 + k] = (half_t)(b1[k] * C1_F * ctau);
  }
  #pragma unroll
  for (int c = 12; c < 32; ++c) fl[c] = (half_t)0.0f;
  #pragma unroll
  for (int c = 6; c < 32; ++c) fg[c] = (half_t)0.0f;
}

// ---------------- weight fusion: Wl = w_l2 @ w_dp_top, Wg = w_g2 @ w_dp_bot ----------------
__global__ void k_fuse(const float* w_l2, const float* b_l2,
                       const float* w_g2, const float* b_g2,
                       const float* w_dp, const float* b_dp,
                       float* Wl, float* Wg, float* bfused)
{
  int i = blockIdx.x * blockDim.x + threadIdx.x;
  if (i < 64*128) {
    int k = i >> 7, n = i & 127;
    float s = 0.0f;
    for (int m = 0; m < 128; ++m) s += w_l2[k*128 + m] * w_dp[m*128 + n];
    Wl[i] = s;
  } else if (i < 2*64*128) {
    int q = i - 64*128;
    int k = q >> 7, n = q & 127;
    float s = 0.0f;
    for (int m = 0; m < 128; ++m) s += w_g2[k*128 + m] * w_dp[(128+m)*128 + n];
    Wg[q] = s;
  } else if (i < 2*64*128 + 128) {
    int n = i - 2*64*128;
    float s = b_dp[n];
    for (int m = 0; m < 128; ++m)
      s += b_l2[m] * w_dp[m*128 + n] + b_g2[m] * w_dp[(128+m)*128 + n];
    bfused[n] = s;
  }
}

// ---------------- pack B operands into WMMA register layout (f16) ----------------
// within a 32x16 f16 B tile: lane = n + 16*h; elem e<8 -> k = 8h+e, e>=8 -> k = 16+8h+(e-8)
__global__ void k_pack(const float* w_l1, const float* w_g1,
                       const float* Wl, const float* Wg,
                       const float* w_in, const float* w_out, half_t* packs)
{
  int gid = blockIdx.x * blockDim.x + threadIdx.x;
  if (gid >= NTILES * 512) return;
  int tile = gid >> 9;
  int q = gid & 511;
  int lane = q >> 4, elem = q & 15;
  int nloc = lane & 15;
  int kloc = (elem < 8) ? ((lane >> 4) * 8 + elem) : (16 + (lane >> 4) * 8 + (elem - 8));   // 0..31
  float v = 0.0f;
  if (tile < 4) {
    int n = tile*16 + nloc;
    if (kloc < 12) v = w_l1[kloc*64 + n];
  } else if (tile < 8) {
    int n = (tile-4)*16 + nloc;
    if (kloc < 6) v = w_g1[kloc*64 + n];
  } else if (tile < 24) {
    int tt = tile - 8; int s = tt >> 3, t = tt & 7;
    v = Wl[(s*32 + kloc)*128 + (t*16 + nloc)];
  } else if (tile < 40) {
    int tt = tile - 24; int s = tt >> 3, t = tt & 7;
    v = Wg[(s*32 + kloc)*128 + (t*16 + nloc)];
  } else if (tile < 72) {
    int tt = tile - 40; int s = tt >> 3, t = tt & 7;
    v = w_in[(s*32 + kloc)*128 + (t*16 + nloc)];
  } else {
    int tt = tile - 72; int s = tt >> 3, t = tt & 7;
    v = w_out[(s*32 + kloc)*128 + (t*16 + nloc)];
  }
  packs[gid] = (half_t)v;
}

// ---------------- node embedding: node_emb = x @ w_in + b_in (WMMA) ----------------
__global__ __launch_bounds__(64) void k_nodeemb(const float* x, const float* b_in,
                                                const half_t* packs, float* node_emb)
{
  __shared__ __align__(32) half_t sx[2][16][128];
  int w = threadIdx.x >> 5, lane = threadIdx.x & 31;
  int tile = blockIdx.x * 2 + w;
  int base = tile * 16;
  int g = lane >> 4, m = lane & 15;
  __builtin_prefetch(packs + (size_t)40*512, 0, 3);

  for (int idx = lane; idx < 16*128; idx += 32) {
    int r = idx >> 7, c = idx & 127;
    sx[w][r][c] = (half_t)x[(size_t)(base + r)*128 + c];
  }
  __syncthreads();

  v8f acc[8];
  #pragma unroll
  for (int t = 0; t < 8; ++t) acc[t] = zero8();

  const v16h* bp = (const v16h*)(packs + (size_t)40*512);
  #pragma unroll
  for (int s = 0; s < 4; ++s) {
    v16h a;
    #pragma unroll
    for (int e2 = 0; e2 < 16; ++e2) a[e2] = sx[w][m][s*32 + amapK(g, e2)];
    #pragma unroll
    for (int t = 0; t < 8; ++t) {
      v16h b = bp[(s*8 + t)*32 + lane];
      acc[t] = __builtin_amdgcn_wmma_f32_16x16x32_f16(false, a, false, b,
                                                      (short)0, acc[t], false, false);
    }
  }
  #pragma unroll
  for (int t = 0; t < 8; ++t) {
    int col = t*16 + m;
    float bias = b_in[col];
    #pragma unroll
    for (int r = 0; r < 8; ++r) {
      int row = base + g*8 + r;
      node_emb[(size_t)row*128 + col] = acc[t][r] + bias;
    }
  }
}

// ---------------- fused edge MLP + message + scatter (WMMA) ----------------
__global__ __launch_bounds__(64) void k_edge(const half_t* featL, const half_t* featG,
                                             const half_t* packs, const float* bfused,
                                             const float* b_l1, const float* b_g1,
                                             const float* node_emb, const float* ef,
                                             const float* ew, const int* el,
                                             const float* deg_in, float* upd)
{
  __shared__ __align__(32) half_t sh[2][16][64];
  int w = threadIdx.x >> 5, lane = threadIdx.x & 31;
  int tile = blockIdx.x * 2 + w;
  int e0 = tile * 16;
  int g = lane >> 4, m = lane & 15;
  const v16h* bp = (const v16h*)packs;
  __builtin_prefetch(packs, 0, 3);

  unsigned lds_base = (unsigned)(size_t)&sh[w][0][0];   // LDS byte offset (low 32 bits)

  v8f acc[8];
  #pragma unroll
  for (int t = 0; t < 8; ++t) acc[t] = zero8();

  // ===== torsion path: feat12 -> 64 -> silu -> (fused 64x128) =====
  {
    // async stage: 1KB tile of featL (16 edges x 32 halves), 2 x b128 per lane
    const char* gsrc = (const char*)featL + (size_t)e0 * 64;
    #pragma unroll
    for (int p = 0; p < 2; ++p) {
      unsigned off16 = (unsigned)(lane + 32*p) * 16u;
      async_g2l_b128(lds_base + off16, (unsigned long long)(gsrc + off16));
    }
    wait_asynccnt0();
    __syncthreads();

    v16h a;
    #pragma unroll
    for (int e2 = 0; e2 < 16; ++e2) a[e2] = sh[w][m][amapK(g, e2)];
    v8f cl[4];
    #pragma unroll
    for (int t = 0; t < 4; ++t) {
      cl[t] = zero8();
      v16h b = bp[(0 + t)*32 + lane];
      cl[t] = __builtin_amdgcn_wmma_f32_16x16x32_f16(false, a, false, b,
                                                     (short)0, cl[t], false, false);
    }
    __syncthreads();
    #pragma unroll
    for (int t = 0; t < 4; ++t) {
      int col = t*16 + m;
      float bias = b_l1[col];
      #pragma unroll
      for (int r8 = 0; r8 < 8; ++r8)
        sh[w][g*8 + r8][col] = (half_t)siluf(cl[t][r8] + bias);
    }
    __syncthreads();
    #pragma unroll
    for (int s = 0; s < 2; ++s) {
      v16h a2;
      #pragma unroll
      for (int e2 = 0; e2 < 16; ++e2) a2[e2] = sh[w][m][s*32 + amapK(g, e2)];
      #pragma unroll
      for (int t = 0; t < 8; ++t) {
        v16h b = bp[(8 + s*8 + t)*32 + lane];
        acc[t] = __builtin_amdgcn_wmma_f32_16x16x32_f16(false, a2, false, b,
                                                        (short)0, acc[t], false, false);
      }
    }
    __syncthreads();
  }

  // ===== angle path: feat6 -> 64 -> silu -> (fused 64x128) =====
  {
    const char* gsrc = (const char*)featG + (size_t)e0 * 64;
    #pragma unroll
    for (int p = 0; p < 2; ++p) {
      unsigned off16 = (unsigned)(lane + 32*p) * 16u;
      async_g2l_b128(lds_base + off16, (unsigned long long)(gsrc + off16));
    }
    wait_asynccnt0();
    __syncthreads();

    v16h a;
    #pragma unroll
    for (int e2 = 0; e2 < 16; ++e2) a[e2] = sh[w][m][amapK(g, e2)];
    v8f cg[4];
    #pragma unroll
    for (int t = 0; t < 4; ++t) {
      cg[t] = zero8();
      v16h b = bp[(4 + t)*32 + lane];
      cg[t] = __builtin_amdgcn_wmma_f32_16x16x32_f16(false, a, false, b,
                                                     (short)0, cg[t], false, false);
    }
    __syncthreads();
    #pragma unroll
    for (int t = 0; t < 4; ++t) {
      int col = t*16 + m;
      float bias = b_g1[col];
      #pragma unroll
      for (int r8 = 0; r8 < 8; ++r8)
        sh[w][g*8 + r8][col] = (half_t)siluf(cg[t][r8] + bias);
    }
    __syncthreads();
    #pragma unroll
    for (int s = 0; s < 2; ++s) {
      v16h a2;
      #pragma unroll
      for (int e2 = 0; e2 < 16; ++e2) a2[e2] = sh[w][m][s*32 + amapK(g, e2)];
      #pragma unroll
      for (int t = 0; t < 8; ++t) {
        v16h b = bp[(24 + s*8 + t)*32 + lane];
        acc[t] = __builtin_amdgcn_wmma_f32_16x16x32_f16(false, a2, false, b,
                                                        (short)0, acc[t], false, false);
      }
    }
  }

  // ===== epilogue: msg = emb + node_emb[ni] + dist; scale; scatter-add =====
  int ni[8], no[8];
  float pre[8], sc[8];
  #pragma unroll
  for (int r8 = 0; r8 < 8; ++r8) {
    int e = e0 + g*8 + r8;
    ni[r8] = el[2*e];
    no[r8] = el[2*e + 1];
    pre[r8] = ef[e];
    sc[r8] = ew[e] / (sqrtf(deg_in[ni[r8]]) + 1e-10f);
  }
  #pragma unroll
  for (int t = 0; t < 8; ++t) {
    int col = t*16 + m;
    float bf = bfused[col];
    #pragma unroll
    for (int r8 = 0; r8 < 8; ++r8) {
      float val = (acc[t][r8] + bf + node_emb[(size_t)ni[r8]*128 + col] + pre[r8]) * sc[r8];
      atomicAdd(&upd[(size_t)no[r8]*128 + col], val);
    }
  }
}

// ---------------- output GEMM: (upd/sqrt(deg_out)) @ w_out + b_out (WMMA) ----------------
__global__ __launch_bounds__(64) void k_out(const float* upd, const float* deg_out,
                                            const float* b_out, const half_t* packs,
                                            float* preBN, float* colsum, float* colsumsq)
{
  __shared__ __align__(32) half_t sx[2][16][128];
  int w = threadIdx.x >> 5, lane = threadIdx.x & 31;
  int tile = blockIdx.x * 2 + w;
  int base = tile * 16;
  int g = lane >> 4, m = lane & 15;
  __builtin_prefetch(packs + (size_t)72*512, 0, 3);

  for (int idx = lane; idx < 16*128; idx += 32) {
    int r = idx >> 7, c = idx & 127;
    float rs = 1.0f / (sqrtf(deg_out[base + r]) + 1e-10f);
    sx[w][r][c] = (half_t)(upd[(size_t)(base + r)*128 + c] * rs);
  }
  __syncthreads();

  v8f acc[8];
  #pragma unroll
  for (int t = 0; t < 8; ++t) acc[t] = zero8();

  const v16h* bp = (const v16h*)(packs + (size_t)72*512);
  #pragma unroll
  for (int s = 0; s < 4; ++s) {
    v16h a;
    #pragma unroll
    for (int e2 = 0; e2 < 16; ++e2) a[e2] = sx[w][m][s*32 + amapK(g, e2)];
    #pragma unroll
    for (int t = 0; t < 8; ++t) {
      v16h b = bp[(s*8 + t)*32 + lane];
      acc[t] = __builtin_amdgcn_wmma_f32_16x16x32_f16(false, a, false, b,
                                                      (short)0, acc[t], false, false);
    }
  }
  #pragma unroll
  for (int t = 0; t < 8; ++t) {
    int col = t*16 + m;
    float bias = b_out[col];
    float csum = 0.0f, csq = 0.0f;
    #pragma unroll
    for (int r = 0; r < 8; ++r) {
      int row = base + g*8 + r;
      float val = acc[t][r] + bias;
      preBN[(size_t)row*128 + col] = val;
      csum += val;
      csq  += val * val;
    }
    atomicAdd(&colsum[col], csum);
    atomicAdd(&colsumsq[col], csq);
  }
}

// ---------------- batchnorm + silu ----------------
__global__ void k_norm(const float* preBN, const float* colsum, const float* colsumsq,
                       const float* gamma, const float* beta, float* out)
{
  int i = blockIdx.x * blockDim.x + threadIdx.x;
  if (i >= NNODE * 128) return;
  int col = i & 127;
  float mean = colsum[col] * (1.0f / NNODE);
  float var = colsumsq[col] * (1.0f / NNODE) - mean * mean;
  float v = (preBN[i] - mean) * rsqrtf(var + 1e-5f) * gamma[col] + beta[col];
  out[i] = siluf(v);
}

// ---------------- launch ----------------
extern "C" void kernel_launch(void* const* d_in, const int* in_sizes, int n_in,
                              void* d_out, int out_size, void* d_ws, size_t ws_size,
                              hipStream_t stream)
{
  const float* x     = (const float*)d_in[0];
  const float* pos   = (const float*)d_in[1];
  const float* ef    = (const float*)d_in[2];
  const float* ew    = (const float*)d_in[3];
  const int*   el    = (const int*)d_in[4];
  const float* w_in  = (const float*)d_in[5];
  const float* b_in  = (const float*)d_in[6];
  const float* w_l1  = (const float*)d_in[7];
  const float* b_l1  = (const float*)d_in[8];
  const float* w_l2  = (const float*)d_in[9];
  const float* b_l2  = (const float*)d_in[10];
  const float* w_g1  = (const float*)d_in[11];
  const float* b_g1  = (const float*)d_in[12];
  const float* w_g2  = (const float*)d_in[13];
  const float* b_g2  = (const float*)d_in[14];
  const float* w_dp  = (const float*)d_in[15];
  const float* b_dp  = (const float*)d_in[16];
  const float* w_out = (const float*)d_in[17];
  const float* b_out = (const float*)d_in[18];
  const float* gamma = (const float*)d_in[19];
  const float* beta  = (const float*)d_in[20];
  float* out = (float*)d_out;

  char* ws = (char*)d_ws;
  size_t off = 0;
  auto salloc = [&](size_t bytes) -> void* {
    void* p = ws + off;
    off = (off + bytes + 255) & ~(size_t)255;
    return p;
  };
  float* deg_in  = (float*)salloc(NNODE * 4);
  float* deg_out = (float*)salloc(NNODE * 4);
  unsigned long long* mo1 = (unsigned long long*)salloc(NNODE * 8);
  unsigned long long* mi1 = (unsigned long long*)salloc(NNODE * 8);
  unsigned long long* mo2 = (unsigned long long*)salloc(NNODE * 8);
  unsigned long long* mi2 = (unsigned long long*)salloc(NNODE * 8);
  float* node_emb = (float*)salloc((size_t)NNODE * 128 * 4);
  float* upd      = (float*)salloc((size_t)NNODE * 128 * 4);
  float* preBN    = (float*)salloc((size_t)NNODE * 128 * 4);
  half_t* featL   = (half_t*)salloc((size_t)NEDGE * 32 * 2);
  half_t* featG   = (half_t*)salloc((size_t)NEDGE * 32 * 2);
  float* fWl      = (float*)salloc(64 * 128 * 4);
  float* fWg      = (float*)salloc(64 * 128 * 4);
  float* bfused   = (float*)salloc(128 * 4);
  half_t* packs   = (half_t*)salloc((size_t)NTILES * 512 * 2);
  float* colsum   = (float*)salloc(128 * 4);
  float* colsumsq = (float*)salloc(128 * 4);

  k_init<<<(NNODE*128 + 255)/256, 256, 0, stream>>>(deg_in, deg_out, mo1, mi1, mo2, mi2,
                                                    upd, colsum, colsumsq);
  k_deg<<<NEDGE/256, 256, 0, stream>>>(el, deg_in, deg_out);
  k_min1<<<NEDGE/256, 256, 0, stream>>>(el, ef, mo1, mi1);
  k_min2<<<NEDGE/256, 256, 0, stream>>>(el, ef, mo1, mi1, mo2, mi2);
  k_geom<<<NEDGE/256, 256, 0, stream>>>(pos, el, ef, mo1, mi1, mo2, mi2, featL, featG);
  k_fuse<<<(2*64*128 + 128 + 255)/256, 256, 0, stream>>>(w_l2, b_l2, w_g2, b_g2,
                                                         w_dp, b_dp, fWl, fWg, bfused);
  k_pack<<<(NTILES*512 + 255)/256, 256, 0, stream>>>(w_l1, w_g1, fWl, fWg, w_in, w_out, packs);
  k_nodeemb<<<NNODE/32, 64, 0, stream>>>(x, b_in, packs, node_emb);
  k_edge<<<NEDGE/32, 64, 0, stream>>>(featL, featG, packs, bfused, b_l1, b_g1,
                                      node_emb, ef, ew, el, deg_in, upd);
  k_out<<<NNODE/32, 64, 0, stream>>>(upd, deg_out, b_out, packs, preBN, colsum, colsumsq);
  k_norm<<<(NNODE*128 + 255)/256, 256, 0, stream>>>(preBN, colsum, colsumsq, gamma, beta, out);
}
